// SelfMatchingAttention_1297080124081
// MI455X (gfx1250) — matrix operand order repaired and checked by reading, not compiled, not run on hardware
//
#include <hip/hip_runtime.h>
#include <hip/hip_bf16.h>

// Shapes
#define P_LEN 512
#define BB    8
#define DD    256
#define HH    75
#define HP    80      // padded H for workspace alignment
#define OO    128
#define TWO_D 512     // 2*D
#define G3O   384     // 3*O
#define ROWS  4096    // P*B

typedef __attribute__((ext_vector_type(16))) _Float16 v16h;
typedef __attribute__((ext_vector_type(8)))  float    v8f;
typedef unsigned int u32x4 __attribute__((ext_vector_type(4)));
typedef int          i32x8 __attribute__((ext_vector_type(8)));
typedef int          i32x4 __attribute__((ext_vector_type(4)));

union FragU { v16h h; unsigned u[8]; };

#if __has_builtin(__builtin_amdgcn_tensor_load_to_lds) && __has_builtin(__builtin_amdgcn_s_wait_tensorcnt)
#define HAVE_TDM 1
#else
#define HAVE_TDM 0
#endif

__device__ __forceinline__ float fast_sigmoid(float x) {
  float e = __builtin_amdgcn_exp2f(-1.44269504089f * x);
  return 1.0f / (1.0f + e);
}
__device__ __forceinline__ float fast_tanh(float x) {
  // 1 - 2/(1+e^{2x}); overflow-safe at both ends
  float e = __builtin_amdgcn_exp2f(2.88539008178f * x);
  return 1.0f - 2.0f / (e + 1.0f);
}

#if HAVE_TDM
// Issue a 1-D TDM copy of `nelem` f32 from global to LDS (wave-level op).
// D# built per CDNA5 ISA sec. 8.3/8.4: group0 = {count/flags, lds_addr,
// global_addr lo, global_addr hi | type=2}; group1 = {data_size=4B,
// tensor_dim0 = tile_dim0 = stride = nelem, tensor_dim1 = 1}; groups 2/3 = 0.
// This toolchain exposes the 6-arg builtin (g0, g1, g2, g3, g4, cpol).
__device__ __forceinline__ void tdm_load_1d(unsigned lds_off, const float* gptr,
                                            unsigned nelem) {
  unsigned long long ga = (unsigned long long)gptr;
  u32x4 g0;
  g0[0] = 1u;                                                // count=1 (user D#)
  g0[1] = lds_off;                                           // lds_addr (bytes)
  g0[2] = (unsigned)(ga & 0xffffffffu);                      // global_addr[31:0]
  g0[3] = (unsigned)((ga >> 32) & 0x01ffffffu) | (2u << 30); // addr[56:32], type=2
  i32x8 g1;
  g1[0] = (int)(2u << 16);                          // workgroup_mask=0, data_size=4B
  g1[1] = (int)((nelem & 0xffffu) << 16);           // tensor_dim0[15:0]   (bits 63:48)
  g1[2] = (int)(((nelem >> 16) & 0xffffu) | (1u << 16)); // td0[31:16], tensor_dim1=1
  g1[3] = (int)((nelem & 0xffffu) << 16);           // td1 hi=0, tile_dim0 (bits 127:112)
  g1[4] = 0;                                        // tile_dim1=0 (unused), tile_dim2=0
  g1[5] = (int)nelem;                               // tensor_dim0_stride[31:0]
  g1[6] = 0;                                        // stride hi, td1_stride lo
  g1[7] = 0;
  i32x4 gz4 = (i32x4){0, 0, 0, 0};
  i32x8 gz8 = (i32x8){0, 0, 0, 0, 0, 0, 0, 0};
  __builtin_amdgcn_tensor_load_to_lds(g0, g1, gz4, gz4, gz8, 0);
}
__device__ __forceinline__ unsigned lds_offset_of(const void* p) {
  // generic LDS address = {aperture[63:32], byte_offset[31:0]} (ISA 10.2)
  return (unsigned)(unsigned long long)p;
}
#endif

// ---------------------------------------------------------------------------
// Kernel 1: Wv1 = v @ Wvp1^T, Wv2 = v @ Wvp2^T   (per (p,b) row; H=75, K=256)
// ---------------------------------------------------------------------------
__global__ void proj_kernel(const float* __restrict__ v,
                            const float* __restrict__ W1,
                            const float* __restrict__ W2,
                            float* __restrict__ wv1, float* __restrict__ wv2) {
  int pb = blockIdx.x;                 // 0..4095
  __shared__ float vr[DD];
  int tid = threadIdx.x;               // 128 threads
  vr[tid]       = v[(size_t)pb * DD + tid];
  vr[tid + 128] = v[(size_t)pb * DD + 128 + tid];
  __syncthreads();
  if (tid < HP) {
    float a1 = 0.f, a2 = 0.f;
    if (tid < HH) {
      const float* w1 = W1 + (size_t)tid * DD;
      const float* w2 = W2 + (size_t)tid * DD;
      for (int d = 0; d < DD; ++d) {
        float x = vr[d];
        a1 += x * w1[d];
        a2 += x * w2[d];
      }
    }
    wv1[(size_t)pb * HP + tid] = a1;
    wv2[(size_t)pb * HP + tid] = a2;
  }
}

// ---------------------------------------------------------------------------
// Kernel 2: generic f32 -> f16 conversion
// ---------------------------------------------------------------------------
__global__ void cvt_kernel(const float* __restrict__ src, _Float16* __restrict__ dst, int n) {
  int i = blockIdx.x * 256 + threadIdx.x;
  if (i < n) dst[i] = (_Float16)src[i];
}

// ---------------------------------------------------------------------------
// Kernel 3: vT16[b][d][i] = f16(v[i][b][d])   (K-contiguous B operand for context GEMM)
// ---------------------------------------------------------------------------
__global__ void vtrans_kernel(const float* __restrict__ v, _Float16* __restrict__ vt) {
  int idx = blockIdx.x * 256 + threadIdx.x;          // 0 .. 8*256*512-1
  int i = idx & (P_LEN - 1);
  int rest = idx >> 9;
  int d = rest & (DD - 1);
  int b = rest >> 8;
  vt[idx] = (_Float16)v[((size_t)i * BB + b) * DD + d];
}

// ---------------------------------------------------------------------------
// Kernel 4: g16 first half: g16[row][k] = f16(v[row][k]), k<256 (row = p*B+b)
// ---------------------------------------------------------------------------
__global__ void gfill_kernel(const float* __restrict__ v, _Float16* __restrict__ g16) {
  int idx = blockIdx.x * 256 + threadIdx.x;          // 0 .. 4096*256-1
  int k = idx & (DD - 1);
  int row = idx >> 8;
  g16[(size_t)row * TWO_D + k] = (_Float16)v[(size_t)row * DD + k];
}

// ---------------------------------------------------------------------------
// Kernel 5: scores + softmax over keys i, per output column (j,b).
// scores[i] = sum_h vw[h]*tanh(Wv2[i,b,h] + Wv1[j,b,h]); a = softmax_i
// writes aT16[(b*512 + j)*512 + i]  (f16, K-contiguous A operand for context GEMM)
// ---------------------------------------------------------------------------
__global__ void score_softmax_kernel(const float* __restrict__ wv1,
                                     const float* __restrict__ wv2,
                                     const float* __restrict__ vw,
                                     _Float16* __restrict__ aT16) {
  int j = blockIdx.x, b = blockIdx.y;
  __shared__ float q[HP];
  __shared__ float w[HP];
  __shared__ float red[256];
  int tid = threadIdx.x;  // 256
  if (tid < HP) {
    q[tid] = wv1[((size_t)j * BB + b) * HP + tid];
    w[tid] = (tid < HH) ? vw[tid] : 0.f;
  }
  __syncthreads();

  float s[2];
#pragma unroll
  for (int u = 0; u < 2; ++u) {
    int i = tid + u * 256;
    const float* krow = wv2 + ((size_t)i * BB + b) * HP;
    float acc = 0.f;
    for (int h = 0; h < HH; ++h) acc += w[h] * fast_tanh(krow[h] + q[h]);
    s[u] = acc;
  }
  // max reduce over 512
  red[tid] = fmaxf(s[0], s[1]);
  __syncthreads();
  for (int off = 128; off > 0; off >>= 1) {
    if (tid < off) red[tid] = fmaxf(red[tid], red[tid + off]);
    __syncthreads();
  }
  float m = red[0];
  __syncthreads();
  float e0 = __builtin_amdgcn_exp2f((s[0] - m) * 1.44269504089f);
  float e1 = __builtin_amdgcn_exp2f((s[1] - m) * 1.44269504089f);
  red[tid] = e0 + e1;
  __syncthreads();
  for (int off = 128; off > 0; off >>= 1) {
    if (tid < off) red[tid] += red[tid + off];
    __syncthreads();
  }
  float inv = 1.0f / red[0];
  size_t base = ((size_t)b * P_LEN + j) * P_LEN;
  aT16[base + tid]       = (_Float16)(e0 * inv);
  aT16[base + tid + 256] = (_Float16)(e1 * inv);
}

// ---------------------------------------------------------------------------
// WMMA GEMM: C(MxN) = A(MxK,f16,row-major) @ Bt(NxK,f16,row-major)^T, f32 accum.
// Wave tile 32x64 (2x4 subtiles of 16x16), block = 8 waves -> 64x256.
// Fragments use contiguous k = 16*half + s for both A and B (wave32).
//   mode 0: context  -> g16[(m*8+z)*512 + 256 + n] = f16(acc)        (grid.z = batch)
//   mode 1: gate     -> gated16[m*512+n] = f16(sigmoid(acc)*g16[m*512+n])
//   mode 2: xp       -> xp[m*384+n] = acc + bias[n]
// ---------------------------------------------------------------------------
__global__ __launch_bounds__(256) void gemm_kernel(
    const _Float16* __restrict__ A, int lda,
    const _Float16* __restrict__ Bt, int ldb,
    int M, int N, int K, int mode,
    _Float16* __restrict__ outh, const _Float16* __restrict__ gsrc,
    float* __restrict__ outf, const float* __restrict__ bias) {
  if (mode == 0) {  // per-batch operand slices
    A  += (size_t)blockIdx.z * P_LEN * P_LEN;
    Bt += (size_t)blockIdx.z * DD * P_LEN;
  }
  int tid  = threadIdx.x;
  int wave = tid >> 5;
  int lane = tid & 31;
  int half = lane >> 4;
  int mr   = lane & 15;
  int wm = wave >> 2, wn = wave & 3;
  int m0 = blockIdx.y * 64 + wm * 32;
  int n0 = blockIdx.x * 256 + wn * 64;
  if (n0 >= N || m0 >= M) return;  // uniform per wave

  v8f acc[2][4];
#pragma unroll
  for (int im = 0; im < 2; ++im)
#pragma unroll
    for (int in = 0; in < 4; ++in) acc[im][in] = (v8f){};

  for (int kk = 0; kk < K; kk += 32) {
    FragU a[2], bfr[4];
#pragma unroll
    for (int im = 0; im < 2; ++im) {
      const _Float16* arow = A + (size_t)(m0 + im * 16 + mr) * lda + kk;
      __builtin_prefetch(arow + 32, 0, 1);
#pragma unroll
      for (int j = 0; j < 8; ++j) {
        int kb = half * 16 + 2 * j;
        a[im].u[j] = *(const unsigned*)(arow + kb);
      }
    }
#pragma unroll
    for (int in = 0; in < 4; ++in) {
      const _Float16* brow = Bt + (size_t)(n0 + in * 16 + mr) * ldb + kk;
#pragma unroll
      for (int j = 0; j < 8; ++j) {
        int kb = half * 16 + 2 * j;
        bfr[in].u[j] = *(const unsigned*)(brow + kb);
      }
    }
#pragma unroll
    for (int im = 0; im < 2; ++im)
#pragma unroll
      for (int in = 0; in < 4; ++in)
        acc[im][in] = __builtin_amdgcn_wmma_f32_16x16x32_f16(
            false, a[im].h, false, bfr[in].h, (short)0, acc[im][in], false, false);
  }

  // epilogue: element r of each acc -> row = m0+im*16+r+8*half, col = n0+in*16+mr
#pragma unroll
  for (int im = 0; im < 2; ++im)
#pragma unroll
    for (int in = 0; in < 4; ++in) {
      int col = n0 + in * 16 + mr;
#pragma unroll
      for (int r = 0; r < 8; ++r) {
        int row = m0 + im * 16 + r + 8 * half;
        float va = acc[im][in][r];
        if (mode == 0) {
          size_t rg = (size_t)row * BB + blockIdx.z;
          outh[rg * TWO_D + DD + col] = (_Float16)va;
        } else if (mode == 1) {
          size_t idx = (size_t)row * TWO_D + col;
          outh[idx] = (_Float16)(fast_sigmoid(va) * (float)gsrc[idx]);
        } else {
          outf[(size_t)row * G3O + col] = va + bias[col];
        }
      }
    }
}

// ---------------------------------------------------------------------------
// Kernel 7: GRU recurrence, one workgroup (1024 thr = 32 waves) per direction.
// Per step: gh = h @ w_hh^T via WMMA (M padded 8->16, h f16 in LDS, w_hh frags
// pre-converted to f16 and held in registers), then elementwise gate math.
// Wave 31 (idle during WMMA) drives a TDM double-buffer: while step t computes,
// the Tensor Data Mover streams step t+1's 12KB xp block into LDS
// (tensor_load_to_lds, ordered per-wave, synced with s_wait_tensorcnt + barrier).
// ---------------------------------------------------------------------------
__global__ __launch_bounds__(1024) void gru_kernel(
    const float* __restrict__ xp_f, const float* __restrict__ xp_b,
    const float* __restrict__ whh_f, const float* __restrict__ whh_b,
    const float* __restrict__ bhh_f, const float* __restrict__ bhh_b,
    float* __restrict__ out) {
  int dir = blockIdx.x;
  const float* xp  = dir ? xp_b  : xp_f;
  const float* whh = dir ? whh_b : whh_f;
  const float* bhh = dir ? bhh_b : bhh_f;
  int col_off = dir ? OO : 0;

  __shared__ __align__(16) _Float16 h16[16][OO];   // rows 8..15 stay zero (M pad)
  __shared__ __align__(16) float    h32[BB][OO];
  __shared__ __align__(16) float    gh[BB][G3O];
#if HAVE_TDM
  __shared__ __align__(16) float    xbuf[2][BB * G3O];  // TDM double buffer (2x12KB)
#endif

  int tid = threadIdx.x;
  int wave = tid >> 5, lane = tid & 31, half = lane >> 4, mr = lane & 15;

  // zero init hidden state (+ pad rows)
  h16[tid >> 7][tid & 127]       = (_Float16)0.f;
  h16[8 + (tid >> 7)][tid & 127] = (_Float16)0.f;
  h32[tid >> 7][tid & 127] = 0.f;

  // preload w_hh^T fragments (waves 0..23 cover N=384), constant across steps
  FragU bfrag[4];
#pragma unroll
  for (int ks = 0; ks < 4; ++ks)
#pragma unroll
    for (int j = 0; j < 8; ++j) bfrag[ks].u[j] = 0u;
  if (wave < 24) {
    const float* wrow = whh + (size_t)(wave * 16 + mr) * OO;
#pragma unroll
    for (int ks = 0; ks < 4; ++ks)
#pragma unroll
      for (int j = 0; j < 8; ++j) {
        int kb = ks * 32 + half * 16 + 2 * j;
        union { unsigned u; _Float16 hh[2]; } pk;
        pk.hh[0] = (_Float16)wrow[kb];
        pk.hh[1] = (_Float16)wrow[kb + 1];
        bfrag[ks].u[j] = pk.u;
      }
  }

#if HAVE_TDM
  // prologue: DMA first timestep's xp block into buffer 0
  if (wave == 31) {
    int t0 = dir ? (P_LEN - 1) : 0;
    tdm_load_1d(lds_offset_of(&xbuf[0][0]), xp + (size_t)t0 * BB * G3O, BB * G3O);
  }
#endif

  int b_ = tid >> 7;       // batch 0..7
  int o  = tid & 127;      // hidden unit
  float bh_r = bhh[o], bh_z = bhh[OO + o], bh_n = bhh[2 * OO + o];
  __syncthreads();

  for (int ts = 0; ts < P_LEN; ++ts) {
    int t = dir ? (P_LEN - 1 - ts) : ts;

    // stage 1: gh = h @ w_hh^T  (WMMA, waves 0..23, uniform branch -> EXEC full)
    if (wave < 24) {
      v8f acc = (v8f){};
#pragma unroll
      for (int ks = 0; ks < 4; ++ks) {
        FragU a;
#pragma unroll
        for (int j = 0; j < 8; ++j) {
          int kb = ks * 32 + half * 16 + 2 * j;
          a.u[j] = *(const unsigned*)&h16[mr][kb];
        }
        acc = __builtin_amdgcn_wmma_f32_16x16x32_f16(
            false, a.h, false, bfrag[ks].h, (short)0, acc, false, false);
      }
      if (half == 0) {  // lanes 0..15 hold valid rows M=0..7
#pragma unroll
        for (int r = 0; r < 8; ++r) gh[r][wave * 16 + mr] = acc[r];
      }
    }
#if HAVE_TDM
    else if (wave == 31) {
      // DMA next block into the other buffer, then ensure current buffer landed.
      // TDM ops from one wave are in-order: tensorcnt<=1 => previous op done.
      if (ts + 1 < P_LEN) {
        int tn = dir ? (P_LEN - 2 - ts) : (ts + 1);
        tdm_load_1d(lds_offset_of(&xbuf[(ts + 1) & 1][0]),
                    xp + (size_t)tn * BB * G3O, BB * G3O);
        __builtin_amdgcn_s_wait_tensorcnt(1);
      } else {
        __builtin_amdgcn_s_wait_tensorcnt(0);
      }
    }
#endif
    __syncthreads();

    // stage 2: elementwise gates + state update (all 1024 threads)
#if HAVE_TDM
    const float* xr = &xbuf[ts & 1][b_ * G3O];
#else
    const float* xr = xp + ((size_t)t * BB + b_) * G3O;
#endif
    float i_r = xr[o], i_z = xr[OO + o], i_n = xr[2 * OO + o];
    float rr = fast_sigmoid(i_r + gh[b_][o] + bh_r);
    float zz = fast_sigmoid(i_z + gh[b_][OO + o] + bh_z);
    float nn = fast_tanh(i_n + rr * (gh[b_][2 * OO + o] + bh_n));
    float hprev = h32[b_][o];
    float hnew = (1.0f - zz) * nn + zz * hprev;
    h32[b_][o] = hnew;
    h16[b_][o] = (_Float16)hnew;
    out[(size_t)t * (BB * 2 * OO) + (size_t)b_ * (2 * OO) + col_off + o] = hnew;
    __syncthreads();
  }
}

// ---------------------------------------------------------------------------
extern "C" void kernel_launch(void* const* d_in, const int* in_sizes, int n_in,
                              void* d_out, int out_size, void* d_ws, size_t ws_size,
                              hipStream_t stream) {
  const float* v      = (const float*)d_in[0];
  const float* Wvp1   = (const float*)d_in[1];
  const float* Wvp2   = (const float*)d_in[2];
  const float* vw     = (const float*)d_in[3];
  const float* Wg     = (const float*)d_in[4];
  const float* w_ih_f = (const float*)d_in[5];
  const float* w_hh_f = (const float*)d_in[6];
  const float* b_ih_f = (const float*)d_in[7];
  const float* b_hh_f = (const float*)d_in[8];
  const float* w_ih_b = (const float*)d_in[9];
  const float* w_hh_b = (const float*)d_in[10];
  const float* b_ih_b = (const float*)d_in[11];
  const float* b_hh_b = (const float*)d_in[12];
  float* out = (float*)d_out;

  // workspace carve-up (bytes)
  char* ws = (char*)d_ws;
  size_t off = 0;
  auto alloc = [&](size_t bytes) { char* p = ws + off; off += (bytes + 255) & ~(size_t)255; return p; };
  float*    wv1     = (float*)   alloc((size_t)ROWS * HP * 4);            // (P*B, 80)
  float*    wv2     = (float*)   alloc((size_t)ROWS * HP * 4);
  _Float16* aT16    = (_Float16*)alloc((size_t)BB * P_LEN * P_LEN * 2);   // (b, j, i)
  _Float16* vT16    = (_Float16*)alloc((size_t)BB * DD * P_LEN * 2);      // (b, d, i)
  _Float16* g16     = (_Float16*)alloc((size_t)ROWS * TWO_D * 2);         // (row, 512)
  _Float16* gated16 = (_Float16*)alloc((size_t)ROWS * TWO_D * 2);
  _Float16* Wg16    = (_Float16*)alloc((size_t)TWO_D * TWO_D * 2);
  _Float16* wihf16  = (_Float16*)alloc((size_t)G3O * TWO_D * 2);
  _Float16* wihb16  = (_Float16*)alloc((size_t)G3O * TWO_D * 2);
  float*    xpf     = (float*)   alloc((size_t)ROWS * G3O * 4);
  float*    xpb     = (float*)   alloc((size_t)ROWS * G3O * 4);

  // 1. input projections for additive attention
  proj_kernel<<<ROWS, 128, 0, stream>>>(v, Wvp1, Wvp2, wv1, wv2);

  // 2. f16 staging of static weights
  cvt_kernel<<<(TWO_D * TWO_D + 255) / 256, 256, 0, stream>>>(Wg, Wg16, TWO_D * TWO_D);
  cvt_kernel<<<(G3O * TWO_D + 255) / 256, 256, 0, stream>>>(w_ih_f, wihf16, G3O * TWO_D);
  cvt_kernel<<<(G3O * TWO_D + 255) / 256, 256, 0, stream>>>(w_ih_b, wihb16, G3O * TWO_D);

  // 3. v transposed per batch (K-contiguous B for context GEMM) + g first half
  vtrans_kernel<<<(BB * DD * P_LEN) / 256, 256, 0, stream>>>(v, vT16);
  gfill_kernel<<<(ROWS * DD) / 256, 256, 0, stream>>>(v, g16);

  // 4. fused scores + softmax -> attention weights
  score_softmax_kernel<<<dim3(P_LEN, BB), 256, 0, stream>>>(wv1, wv2, vw, aT16);

  // 5. context GEMM: c[b] = a[b]^T @ v[b]  (M=512, N=256, K=512), fills g[:,256:]
  gemm_kernel<<<dim3(1, P_LEN / 64, BB), 256, 0, stream>>>(
      aT16, P_LEN, vT16, P_LEN, P_LEN, DD, P_LEN, 0, g16, nullptr, nullptr, nullptr);

  // 6. gate GEMM: gated = sigmoid(g @ Wg^T) * g  (M=4096, N=512, K=512)
  gemm_kernel<<<dim3(TWO_D / 256, ROWS / 64), 256, 0, stream>>>(
      g16, TWO_D, Wg16, TWO_D, ROWS, TWO_D, TWO_D, 1, gated16, g16, nullptr, nullptr);

  // 7. GRU input projections: xp = gated @ w_ih^T + b_ih (M=4096, N=384, K=512)
  gemm_kernel<<<dim3(2, ROWS / 64), 256, 0, stream>>>(
      gated16, TWO_D, wihf16, TWO_D, ROWS, G3O, TWO_D, 2, nullptr, nullptr, xpf, b_ih_f);
  gemm_kernel<<<dim3(2, ROWS / 64), 256, 0, stream>>>(
      gated16, TWO_D, wihb16, TWO_D, ROWS, G3O, TWO_D, 2, nullptr, nullptr, xpb, b_ih_b);

  // 8. bidirectional GRU recurrence (2 concurrent workgroups, WMMA per step,
  //    TDM-prefetched xp double buffer)
  gru_kernel<<<2, 1024, 0, stream>>>(xpf, xpb, w_hh_f, w_hh_b, b_hh_f, b_hh_b, out);
}
